// ExtractorMLP_11759620456600
// MI455X (gfx1250) — compile-verified, matrix operand order checked by reading, not run on hardware
//
#include <hip/hip_runtime.h>

// ExtractorMLP edge-scoring kernel for gfx1250 (MI455X).
// emb: [50000,128] f32 | edge_index: [2,E] int32 | W1:[256,512] b1:[512]
// W2:[512,128] b2:[128] | W3:[128,2] b3:[2]  -> out [E,2] f32
// Strategy: bf16 WMMA (v_wmma_f32_16x16x32_bf16) for layers 1&2 with f32
// accumulate; tiny layer 3 (N=2) in scalar f32 FMA.
// d_ws layout: W1T bf16 [512][256] (256KB) | W2T bf16 [128][512] (128KB).

typedef __attribute__((ext_vector_type(16))) __bf16 v16bf;
typedef __attribute__((ext_vector_type(8)))  float  v8f;

union BFrag { uint4 q[2]; v16bf v; };

__device__ __forceinline__ unsigned short f2bf(float f) {
  unsigned u = __float_as_uint(f);
  u = u + 0x7FFFu + ((u >> 16) & 1u);   // round-to-nearest-even
  return (unsigned short)(u >> 16);
}

__device__ __forceinline__ unsigned int packbf(float lo, float hi) {
  unsigned a = __float_as_uint(lo);
  unsigned b = __float_as_uint(hi);
  a = (a + 0x7FFFu + ((a >> 16) & 1u)) >> 16;
  b = (b + 0x7FFFu + ((b >> 16) & 1u)) >> 16;
  return a | (b << 16);
}

// ---------------- prep: f32 weights -> transposed bf16 in workspace --------
__global__ __launch_bounds__(256) void prep_weights(
    const float* __restrict__ W1, const float* __restrict__ W2,
    unsigned short* __restrict__ W1T, unsigned short* __restrict__ W2T) {
  int i = blockIdx.x * 256 + threadIdx.x;
  if (i < 256 * 512) {              // W1T[n][k], n<512, k<256
    int n = i >> 8, k = i & 255;
    W1T[i] = f2bf(W1[k * 512 + n]);
  }
  if (i < 128 * 512) {              // W2T[n][k], n<128, k<512
    int n = i >> 9, k = i & 511;
    W2T[i] = f2bf(W2[k * 128 + n]);
  }
}

// ---------------- main fused kernel ---------------------------------------
#define MBLK 32   // edges per block (2 WMMA M-tiles)

__global__ __launch_bounds__(256) void extractor_mlp_kernel(
    const float* __restrict__ emb, const int* __restrict__ eidx,
    const unsigned short* __restrict__ W1T, const unsigned short* __restrict__ W2T,
    const float* __restrict__ b1, const float* __restrict__ b2,
    const float* __restrict__ W3, const float* __restrict__ b3,
    float* __restrict__ out, int E) {
  // A1: concat inputs [32 rows][256 K] bf16 (16KB)
  // A2: layer-1 activations [32][512] bf16 (32KB)
  // H2: layer-2 activations [32][128] f32 (16KB)
  __shared__ __align__(16) unsigned short A1[32 * 256];
  __shared__ __align__(16) unsigned short A2[32 * 512];
  __shared__ __align__(16) float          H2[32 * 128];

  const int t    = threadIdx.x;
  const int base = blockIdx.x * MBLK;

  const char* w1b = reinterpret_cast<const char*>(W1T);
  const char* w2b = reinterpret_cast<const char*>(W2T);

  // Hint the L2-resident bf16 weights toward this WGP (global_prefetch_b8).
  __builtin_prefetch(w1b + (size_t)t * 1024, 0, 1);
  __builtin_prefetch(w2b + (size_t)t * 512, 0, 1);

  // ---- Phase 0: gather endpoint embeddings, f32->bf16, stage in LDS ------
  {
    int row = t >> 3;              // edge within block (0..31)
    int g   = t & 7;               // 32-column group (0..7); g<4 -> col half
    int e   = base + row; if (e >= E) e = E - 1;
    int src = (g < 4) ? eidx[e] : eidx[E + e];
    const float4* s =
        reinterpret_cast<const float4*>(emb + (size_t)src * 128 + (g & 3) * 32);
    uint4* dst = reinterpret_cast<uint4*>(
        reinterpret_cast<char*>(A1) + row * 512 + g * 64);
#pragma unroll
    for (int i = 0; i < 4; ++i) {
      float4 x = s[2 * i], y = s[2 * i + 1];
      uint4 o;
      o.x = packbf(x.x, x.y); o.y = packbf(x.z, x.w);
      o.z = packbf(y.x, y.y); o.w = packbf(y.z, y.w);
      dst[i] = o;
    }
  }
  __syncthreads();

  const int w    = t >> 5;       // wave id (0..7), wave32
  const int lane = t & 31;
  const int h    = lane >> 4;    // K-half select in 16-bit fragment layout
  const int ln   = lane & 15;    // row/col within tile

  // ---- Phase 1: [32,256] @ W1 -> relu -> A2 [32,512] bf16 ----------------
  {
    v8f acc[2][4];
#pragma unroll
    for (int mt = 0; mt < 2; ++mt)
#pragma unroll
      for (int nt = 0; nt < 4; ++nt)
        acc[mt][nt] = (v8f){0.f, 0.f, 0.f, 0.f, 0.f, 0.f, 0.f, 0.f};

    const char* a1b = reinterpret_cast<const char*>(A1);
    for (int ks = 0; ks < 8; ++ks) {        // K = 8 * 32 = 256
      const int kb = ks * 64 + h * 16;      // byte offset of lane's K-run
      BFrag a[2], b[4];
#pragma unroll
      for (int mt = 0; mt < 2; ++mt) {
        const char* p = a1b + (mt * 16 + ln) * 512 + kb;
        a[mt].q[0] = *reinterpret_cast<const uint4*>(p);
        a[mt].q[1] = *reinterpret_cast<const uint4*>(p + 32);
      }
#pragma unroll
      for (int nt = 0; nt < 4; ++nt) {
        const char* p = w1b + (size_t)(w * 64 + nt * 16 + ln) * 512 + kb;
        b[nt].q[0] = *reinterpret_cast<const uint4*>(p);
        b[nt].q[1] = *reinterpret_cast<const uint4*>(p + 32);
      }
#pragma unroll
      for (int mt = 0; mt < 2; ++mt)
#pragma unroll
        for (int nt = 0; nt < 4; ++nt)
          acc[mt][nt] = __builtin_amdgcn_wmma_f32_16x16x32_bf16(
              false, a[mt].v, false, b[nt].v, (short)0, acc[mt][nt],
              false, false);
    }
    // epilogue: bias + relu + bf16 -> A2
#pragma unroll
    for (int nt = 0; nt < 4; ++nt) {
      const int   n    = w * 64 + nt * 16 + ln;
      const float bias = b1[n];
#pragma unroll
      for (int mt = 0; mt < 2; ++mt)
#pragma unroll
        for (int r = 0; r < 8; ++r) {
          float v = acc[mt][nt][r] + bias;
          v = v > 0.f ? v : 0.f;
          const int M = mt * 16 + h * 8 + r;      // C layout: M = r + 8*(L/16)
          A2[M * 512 + n] = f2bf(v);
        }
    }
  }
  __syncthreads();

  // ---- Phase 2: A2 @ W2 -> relu -> H2 [32,128] f32 -----------------------
  {
    v8f acc[2];
    acc[0] = (v8f){0.f, 0.f, 0.f, 0.f, 0.f, 0.f, 0.f, 0.f};
    acc[1] = acc[0];
    const char* a2b = reinterpret_cast<const char*>(A2);
    const int   n   = w * 16 + ln;          // wave w owns N-tile w
    for (int ks = 0; ks < 16; ++ks) {       // K = 16 * 32 = 512
      const int kb = ks * 64 + h * 16;
      BFrag a0, a1, bb;
      {
        const char* p = a2b + (0 * 16 + ln) * 1024 + kb;
        a0.q[0] = *reinterpret_cast<const uint4*>(p);
        a0.q[1] = *reinterpret_cast<const uint4*>(p + 32);
      }
      {
        const char* p = a2b + (1 * 16 + ln) * 1024 + kb;
        a1.q[0] = *reinterpret_cast<const uint4*>(p);
        a1.q[1] = *reinterpret_cast<const uint4*>(p + 32);
      }
      {
        const char* p = w2b + (size_t)n * 1024 + kb;
        bb.q[0] = *reinterpret_cast<const uint4*>(p);
        bb.q[1] = *reinterpret_cast<const uint4*>(p + 32);
      }
      acc[0] = __builtin_amdgcn_wmma_f32_16x16x32_bf16(
          false, a0.v, false, bb.v, (short)0, acc[0], false, false);
      acc[1] = __builtin_amdgcn_wmma_f32_16x16x32_bf16(
          false, a1.v, false, bb.v, (short)0, acc[1], false, false);
    }
    const float bias = b2[n];
#pragma unroll
    for (int mt = 0; mt < 2; ++mt)
#pragma unroll
      for (int r = 0; r < 8; ++r) {
        float v = acc[mt][r] + bias;
        v = v > 0.f ? v : 0.f;
        const int M = mt * 16 + h * 8 + r;
        H2[M * 128 + n] = v;
      }
  }
  __syncthreads();

  // ---- Phase 3: H2 @ W3 + b3 -> out [32,2] (K=128, N=2: scalar FMA) ------
  if (t < 64) {
    const int e  = t >> 1;
    const int oc = t & 1;
    float acc = b3[oc];
    const float* hrow = &H2[e * 128];
#pragma unroll 4
    for (int k = 0; k < 128; ++k)
      acc = fmaf(hrow[k], W3[k * 2 + oc], acc);
    const int ge = base + e;
    if (ge < E) out[ge * 2 + oc] = acc;
  }
}

extern "C" void kernel_launch(void* const* d_in, const int* in_sizes, int n_in,
                              void* d_out, int out_size, void* d_ws, size_t ws_size,
                              hipStream_t stream) {
  (void)n_in; (void)out_size; (void)ws_size;
  const float* emb = (const float*)d_in[0];
  const int*   eix = (const int*)d_in[1];   // edge_index [2,E] (int32)
  const float* W1  = (const float*)d_in[2];
  const float* b1  = (const float*)d_in[3];
  const float* W2  = (const float*)d_in[4];
  const float* b2  = (const float*)d_in[5];
  const float* W3  = (const float*)d_in[6];
  const float* b3  = (const float*)d_in[7];
  float*       out = (float*)d_out;

  unsigned short* W1T = (unsigned short*)d_ws;            // 512*256 bf16
  unsigned short* W2T = W1T + 512 * 256;                  // 128*512 bf16

  const int E = in_sizes[1] / 2;                          // 800000

  prep_weights<<<512, 256, 0, stream>>>(W1, W2, W1T, W2T);
  extractor_mlp_kernel<<<(E + MBLK - 1) / MBLK, 256, 0, stream>>>(
      emb, eix, W1T, W2T, b1, b2, W3, b3, out, E);
}
